// GPT4TS_37194416783448
// MI455X (gfx1250) — matrix operand order repaired and checked by reading, not compiled, not run on hardware
//
#include <hip/hip_runtime.h>
#include <hip/hip_bf16.h>
#include <stdint.h>

typedef unsigned short u16;
typedef __attribute__((ext_vector_type(16))) __bf16 v16bf;
typedef __attribute__((ext_vector_type(8)))  float  v8f;

#define B_      64
#define L_      512
#define D_      768
#define NH_     12
#define HD_     64
#define TOK_    4
#define PRED_   96
#define NL_     6
#define PATCH_  16
#define STRIDE_ 8
#define SEQ_    68       // TOK + PATCH_NUM(64)
#define NSER_   192      // 3*B
#define ROWS_   (NSER_*SEQ_)   // 13056

// ---------- helpers ----------
__device__ __forceinline__ u16 f2b(float f) {
    unsigned u = __float_as_uint(f);
    unsigned r = u + 0x7fffu + ((u >> 16) & 1u);   // round-to-nearest-even
    return (u16)(r >> 16);
}
__device__ __forceinline__ float gelu_tanh(float x) {
    float x3 = x * x * x;
    float t  = tanhf(0.7978845608028654f * (x + 0.044715f * x3));
    return 0.5f * x * (1.0f + t);
}

// ---------- f32 -> bf16 convert ----------
__global__ void f2b_k(const float* __restrict__ in, u16* __restrict__ out, int n) {
    int i = blockIdx.x * blockDim.x + threadIdx.x;
    int s = gridDim.x * blockDim.x;
    for (; i < n; i += s) out[i] = f2b(in[i]);
}

// ---------- decomposition + per-series normalization ----------
__global__ __launch_bounds__(256) void decomp_k(
    const float* __restrict__ x, float* __restrict__ Z,
    float* __restrict__ means, float* __restrict__ stdevs)
{
    __shared__ float xs[L_], tr[L_], se[L_];
    __shared__ float red[256];
    const int b = blockIdx.x, tid = threadIdx.x;
    const float* xb = x + (size_t)b * L_;
    xs[tid] = xb[tid]; xs[tid + 256] = xb[tid + 256];
    __syncthreads();
    // trend: moving average k=25 with edge padding
    #pragma unroll
    for (int p = 0; p < 2; ++p) {
        int t = tid + p * 256;
        float s = 0.f;
        for (int d = -12; d <= 12; ++d) {
            int j = t + d; j = j < 0 ? 0 : (j > 511 ? 511 : j);
            s += xs[j];
        }
        tr[t] = s * (1.f / 25.f);
    }
    __syncthreads();
    // season: Gaussian smoothing, SPAN=0.5 -> w(d)=exp(-2 d^2), radius 8 is fp32-exact
    #pragma unroll
    for (int p = 0; p < 2; ++p) {
        int t = tid + p * 256;
        float num = 0.f, den = 0.f;
        for (int d = -8; d <= 8; ++d) {
            int j = t + d;
            if (j >= 0 && j < L_) {
                float w = expf(-2.0f * (float)(d * d));
                den += w;
                num += w * (xs[j] - tr[j]);
            }
        }
        se[t] = num / den;
    }
    __syncthreads();
    // per-channel normalization: c=0 trend, c=1 season, c=2 residual
    for (int c = 0; c < 3; ++c) {
        int t0 = tid, t1 = tid + 256;
        float v0 = (c == 0) ? tr[t0] : (c == 1) ? se[t0] : (xs[t0] - tr[t0] - se[t0]);
        float v1 = (c == 0) ? tr[t1] : (c == 1) ? se[t1] : (xs[t1] - tr[t1] - se[t1]);
        red[tid] = v0 + v1; __syncthreads();
        for (int o = 128; o > 0; o >>= 1) { if (tid < o) red[tid] += red[tid + o]; __syncthreads(); }
        float mu = red[0] * (1.f / 512.f);
        __syncthreads();
        float d0 = v0 - mu, d1 = v1 - mu;
        red[tid] = d0 * d0 + d1 * d1; __syncthreads();
        for (int o = 128; o > 0; o >>= 1) { if (tid < o) red[tid] += red[tid + o]; __syncthreads(); }
        float sd = sqrtf(red[0] * (1.f / 512.f) + 1e-5f);
        __syncthreads();
        float inv = 1.f / sd;
        int ser = c * B_ + b;
        Z[(size_t)ser * L_ + t0] = d0 * inv;
        Z[(size_t)ser * L_ + t1] = d1 * inv;
        if (tid == 0) { means[ser] = mu; stdevs[ser] = sd; }
    }
}

// ---------- patch embedding + prompt + positional ----------
__global__ __launch_bounds__(256) void embed_k(
    const float* __restrict__ Z, const float* __restrict__ prompt,
    const float* __restrict__ w_in, const float* __restrict__ b_in,
    const float* __restrict__ wpe, float* __restrict__ H)
{
    const int sIdx = blockIdx.x;   // 0..191
    const int t    = blockIdx.y;   // 0..67
    const int tid  = threadIdx.x;
    float* hrow = H + ((size_t)sIdx * SEQ_ + t) * D_;
    #pragma unroll
    for (int p = 0; p < 3; ++p) {
        int c = tid + p * 256;
        float val;
        if (t < TOK_) {
            val = prompt[((size_t)(sIdx / B_) * TOK_ + t) * D_ + c];
        } else {
            int pa = t - TOK_;
            float acc = b_in[c];
            #pragma unroll
            for (int j = 0; j < PATCH_; ++j) {
                int zi = pa * STRIDE_ + j; if (zi > 511) zi = 511;  // last value repeated
                acc += Z[(size_t)sIdx * L_ + zi] * w_in[j * D_ + c];
            }
            val = acc;
        }
        hrow[c] = val + wpe[(size_t)t * D_ + c];
    }
}

// ---------- LayerNorm (f32 in -> bf16 out) ----------
__global__ __launch_bounds__(256) void ln_k(
    const float* __restrict__ X, const float* __restrict__ g, const float* __restrict__ b,
    u16* __restrict__ out)
{
    __shared__ float red[256];
    const int r = blockIdx.x, tid = threadIdx.x;
    const float* x = X + (size_t)r * D_;
    float a0 = x[tid], a1 = x[tid + 256], a2 = x[tid + 512];
    red[tid] = a0 + a1 + a2; __syncthreads();
    for (int o = 128; o > 0; o >>= 1) { if (tid < o) red[tid] += red[tid + o]; __syncthreads(); }
    float mu = red[0] * (1.f / 768.f);
    __syncthreads();
    float d0 = a0 - mu, d1 = a1 - mu, d2 = a2 - mu;
    red[tid] = d0 * d0 + d1 * d1 + d2 * d2; __syncthreads();
    for (int o = 128; o > 0; o >>= 1) { if (tid < o) red[tid] += red[tid + o]; __syncthreads(); }
    float rs = rsqrtf(red[0] * (1.f / 768.f) + 1e-5f);
    u16* o_ = out + (size_t)r * D_;
    o_[tid]       = f2b(d0 * rs * g[tid]       + b[tid]);
    o_[tid + 256] = f2b(d1 * rs * g[tid + 256] + b[tid + 256]);
    o_[tid + 512] = f2b(d2 * rs * g[tid + 512] + b[tid + 512]);
}

// ---------- WMMA bf16 GEMM: C = A(MxK) * B(KxN) + bias [+res] [gelu] ----------
// Block tile 128x128, BK=64, 8 waves; each wave: 32 rows x 64 cols
// = 2 M-tiles x 4 N-tiles = 8 accumulators, 16 WMMAs per staging round.
template<int RES, int ACT, int OUTB>
__global__ __launch_bounds__(256) void gemm_bf16(
    const u16* __restrict__ A, const u16* __restrict__ Bm,
    const float* __restrict__ bias, const float* __restrict__ res,
    float* __restrict__ outF, u16* __restrict__ outB,
    int M, int N, int K)
{
    __shared__ u16 lA[128 * 72];   // A tile 128x64, padded stride 72 (144B, 16B-aligned)
    __shared__ u16 lB[128 * 72];   // B tile transposed: 128(n) x 64(k), stride 72
    const int tid  = threadIdx.x;
    const int lane = tid & 31, wave = tid >> 5;
    const int half = lane >> 4, lr = lane & 15;
    const int mi2  = wave & 3;          // 32-row band
    const int njb  = (wave >> 2) * 4;   // 4 consecutive 16-col tiles
    const int row0 = blockIdx.x * 128;
    const int n0   = blockIdx.y * 128;

    v8f vzero = {0.f, 0.f, 0.f, 0.f, 0.f, 0.f, 0.f, 0.f};
    v8f acc[2][4];
    #pragma unroll
    for (int mt = 0; mt < 2; ++mt)
        #pragma unroll
        for (int j = 0; j < 4; ++j) acc[mt][j] = vzero;

    for (int k0 = 0; k0 < K; k0 += 64) {
        // stage A (128 rows x 64 k, 8 bf16 per chunk, 4 chunks/thread)
        #pragma unroll
        for (int c = tid; c < 1024; c += 256) {
            int r = c >> 3, ak = (c & 7) * 8;
            *(uint4*)&lA[r * 72 + ak] = *(const uint4*)&A[(size_t)(row0 + r) * K + k0 + ak];
        }
        // stage B transposed (64 k x 128 n, scatter 8 bf16 per chunk)
        #pragma unroll
        for (int c = tid; c < 1024; c += 256) {
            int kk = c >> 4, no = (c & 15) * 8;
            uint4 bv = *(const uint4*)&Bm[(size_t)(k0 + kk) * N + n0 + no];
            const u16* bs = (const u16*)&bv;
            #pragma unroll
            for (int i = 0; i < 8; ++i) lB[(no + i) * 72 + kk] = bs[i];
        }
        if (k0 + 64 < K) {  // global_prefetch_b8 next K-tile
            __builtin_prefetch(&A[(size_t)(row0 + (tid >> 3)) * K + k0 + 64 + (tid & 7) * 8], 0, 0);
            __builtin_prefetch(&Bm[(size_t)(k0 + 64 + (tid >> 4)) * N + n0 + (tid & 15) * 8], 0, 0);
        }
        __syncthreads();
        union Frag { uint4 u[2]; u16 s[16]; v16bf v; };
        #pragma unroll
        for (int kb = 0; kb < 2; ++kb) {
            // A frags: lanes 0-15 -> K {0..7,16..23}, lanes 16-31 -> K {8..15,24..31}
            Frag af[2];
            #pragma unroll
            for (int mt = 0; mt < 2; ++mt) {
                const u16* ap = &lA[(mi2 * 32 + mt * 16 + lr) * 72 + kb * 32 + half * 8];
                af[mt].u[0] = *(const uint4*)ap;
                af[mt].u[1] = *(const uint4*)(ap + 16);
            }
            #pragma unroll
            for (int j = 0; j < 4; ++j) {
                // B frag: lane = column, lanes 0-15 K {0..7,16..23}, lanes 16-31 K {8..15,24..31}
                Frag bf;
                const u16* bp = &lB[((njb + j) * 16 + lr) * 72 + kb * 32 + half * 8];
                bf.u[0] = *(const uint4*)bp;
                bf.u[1] = *(const uint4*)(bp + 16);
                #pragma unroll
                for (int mt = 0; mt < 2; ++mt)
                    acc[mt][j] = __builtin_amdgcn_wmma_f32_16x16x32_bf16(
                        false, af[mt].v, false, bf.v, (short)0, acc[mt][j], false, false);
            }
        }
        __syncthreads();
    }
    // epilogue: C/D layout lane=n, VGPR v -> row half*8+v
    #pragma unroll
    for (int mt = 0; mt < 2; ++mt) {
        #pragma unroll
        for (int j = 0; j < 4; ++j) {
            int col = n0 + (njb + j) * 16 + lr;
            #pragma unroll
            for (int v = 0; v < 8; ++v) {
                int row = row0 + mi2 * 32 + mt * 16 + half * 8 + v;
                size_t idx = (size_t)row * N + col;
                float val = acc[mt][j][v] + bias[col];
                if (RES)  val += res[idx];
                if (ACT)  val = gelu_tanh(val);
                if (OUTB) outB[idx] = f2b(val);
                else      outF[idx] = val;
            }
        }
    }
}

// ---------- attention (per sequence, per head) ----------
__global__ __launch_bounds__(256) void attention_k(
    const float* __restrict__ qkv, u16* __restrict__ outB)
{
    __shared__ float Qs[SEQ_ * HD_], Ks[SEQ_ * HD_], Vs[SEQ_ * HD_];
    __shared__ float Ss[SEQ_ * 72];
    const int s = blockIdx.x, hd = blockIdx.y, tid = threadIdx.x;
    const float* base = qkv + (size_t)s * SEQ_ * (3 * D_);
    for (int idx = tid; idx < SEQ_ * HD_; idx += 256) {
        int i = idx >> 6, d = idx & 63;
        size_t o = (size_t)i * (3 * D_) + hd * HD_ + d;
        Qs[idx] = base[o];
        Ks[idx] = base[o + D_];
        Vs[idx] = base[o + 2 * D_];
    }
    __syncthreads();
    // scores with causal mask, scale 1/sqrt(64)
    for (int idx = tid; idx < SEQ_ * SEQ_; idx += 256) {
        int i = idx / SEQ_, j = idx - i * SEQ_;
        float v = -3.0e38f;
        if (j <= i) {
            float a = 0.f;
            const float* qp = &Qs[i * HD_];
            const float* kp = &Ks[j * HD_];
            #pragma unroll 8
            for (int d = 0; d < HD_; ++d) a += qp[d] * kp[d];
            v = a * 0.125f;
        }
        Ss[i * 72 + j] = v;
    }
    __syncthreads();
    // row softmax: one wave per row
    const int lane = tid & 31, wave = tid >> 5;
    for (int i = wave; i < SEQ_; i += 8) {
        float v0 = Ss[i * 72 + lane];
        float v1 = (lane + 32 < SEQ_) ? Ss[i * 72 + lane + 32] : -3.0e38f;
        float v2 = (lane + 64 < SEQ_) ? Ss[i * 72 + lane + 64] : -3.0e38f;
        float m = fmaxf(v0, fmaxf(v1, v2));
        for (int o = 16; o > 0; o >>= 1) m = fmaxf(m, __shfl_xor(m, o, 32));
        float e0 = (lane <= i) ? expf(v0 - m) : 0.f;
        float e1 = (lane + 32 <= i) ? expf(v1 - m) : 0.f;
        float e2 = (lane + 64 <= i && lane + 64 < SEQ_) ? expf(v2 - m) : 0.f;
        float sum = e0 + e1 + e2;
        for (int o = 16; o > 0; o >>= 1) sum += __shfl_xor(sum, o, 32);
        float inv = 1.f / sum;
        Ss[i * 72 + lane] = e0 * inv;
        if (lane + 32 < SEQ_) Ss[i * 72 + lane + 32] = e1 * inv;
        if (lane + 64 < SEQ_) Ss[i * 72 + lane + 64] = e2 * inv;
    }
    __syncthreads();
    // O = S @ V -> bf16
    for (int idx = tid; idx < SEQ_ * HD_; idx += 256) {
        int i = idx >> 6, d = idx & 63;
        float a = 0.f;
        const float* sp = &Ss[i * 72];
        for (int j = 0; j <= i; ++j) a += sp[j] * Vs[j * HD_ + d];
        outB[(size_t)(s * SEQ_ + i) * D_ + hd * HD_ + d] = f2b(a);
    }
}

// ---------- head GEMM: (192 x 52224) @ (52224 x 96), split-K=51, 1 wave/block ----------
__global__ __launch_bounds__(32) void head_gemm(
    const u16* __restrict__ A, const u16* __restrict__ Bm, float* __restrict__ part)
{
    const int lane = threadIdx.x & 31;
    const int half = lane >> 4, lr = lane & 15;
    const int m0 = blockIdx.x * 16, n0 = blockIdx.y * 16;
    const int kc = blockIdx.z * 1024;
    const int KTOT = SEQ_ * D_;  // 52224
    v8f acc = {0.f, 0.f, 0.f, 0.f, 0.f, 0.f, 0.f, 0.f};
    union Frag { uint4 u[2]; u16 s[16]; v16bf v; } af, bf;
    for (int st = 0; st < 32; ++st) {
        int k0 = kc + st * 32;
        const u16* ap = &A[(size_t)(m0 + lr) * KTOT + k0 + half * 8];
        af.u[0] = *(const uint4*)ap;
        af.u[1] = *(const uint4*)(ap + 16);
        #pragma unroll
        for (int i = 0; i < 16; ++i)
            bf.s[i] = Bm[(size_t)(k0 + half * 8 + (i & 7) + (i >> 3) * 16) * PRED_ + n0 + lr];
        acc = __builtin_amdgcn_wmma_f32_16x16x32_bf16(
            false, af.v, false, bf.v, (short)0, acc, false, false);
    }
    float* p = part + (size_t)blockIdx.z * (NSER_ * PRED_);
    #pragma unroll
    for (int v = 0; v < 8; ++v)
        p[(m0 + half * 8 + v) * PRED_ + n0 + lr] = acc[v];
}

// ---------- deterministic reduce + rescale + channel sum ----------
__global__ __launch_bounds__(96) void combine_k(
    const float* __restrict__ part, const float* __restrict__ b_out,
    const float* __restrict__ means, const float* __restrict__ stdevs,
    float* __restrict__ out)
{
    const int b = blockIdx.x, p = threadIdx.x;
    float tot = 0.f;
    for (int c = 0; c < 3; ++c) {
        int s = c * B_ + b;
        float acc = b_out[p];
        for (int z = 0; z < 51; ++z)
            acc += part[(size_t)z * (NSER_ * PRED_) + s * PRED_ + p];
        tot += acc * stdevs[s] + means[s];
    }
    out[b * PRED_ + p] = tot;
}

// ---------- host ----------
extern "C" void kernel_launch(void* const* d_in, const int* in_sizes, int n_in,
                              void* d_out, int out_size, void* d_ws, size_t ws_size,
                              hipStream_t stream)
{
    (void)in_sizes; (void)n_in; (void)out_size; (void)ws_size;
    const float* x      = (const float*)d_in[0];
    const float* prompt = (const float*)d_in[1];
    const float* w_in   = (const float*)d_in[2];
    const float* b_in   = (const float*)d_in[3];
    const float* w_out  = (const float*)d_in[4];
    const float* b_out  = (const float*)d_in[5];
    const float* wpe    = (const float*)d_in[6];
    const float* ln1_g  = (const float*)d_in[7];
    const float* ln1_b  = (const float*)d_in[8];
    const float* w_qkv  = (const float*)d_in[9];
    const float* b_qkv  = (const float*)d_in[10];
    const float* w_ao   = (const float*)d_in[11];
    const float* b_ao   = (const float*)d_in[12];
    const float* ln2_g  = (const float*)d_in[13];
    const float* ln2_b  = (const float*)d_in[14];
    const float* w_fc   = (const float*)d_in[15];
    const float* b_fc   = (const float*)d_in[16];
    const float* w_mo   = (const float*)d_in[17];
    const float* b_mo   = (const float*)d_in[18];
    const float* lnf_g  = (const float*)d_in[19];
    const float* lnf_b  = (const float*)d_in[20];

    char* ws = (char*)d_ws;
    size_t off = 0;
    auto alloc = [&](size_t bytes) -> void* {
        void* p = ws + off;
        off = (off + bytes + 255) & ~(size_t)255;
        return p;
    };
    u16*   WQKV  = (u16*)  alloc((size_t)NL_ * D_ * 3 * D_ * 2);
    u16*   WAO   = (u16*)  alloc((size_t)NL_ * D_ * D_ * 2);
    u16*   WFC   = (u16*)  alloc((size_t)NL_ * D_ * 4 * D_ * 2);
    u16*   WMO   = (u16*)  alloc((size_t)NL_ * 4 * D_ * D_ * 2);
    u16*   WOUT  = (u16*)  alloc((size_t)SEQ_ * D_ * PRED_ * 2);
    float* Z     = (float*)alloc((size_t)NSER_ * L_ * 4);
    float* MEANS = (float*)alloc((size_t)NSER_ * 4);
    float* STDEV = (float*)alloc((size_t)NSER_ * 4);
    float* H     = (float*)alloc((size_t)ROWS_ * D_ * 4);
    u16*   ABUF  = (u16*)  alloc((size_t)ROWS_ * D_ * 2);
    float* QKV   = (float*)alloc((size_t)ROWS_ * 3 * D_ * 4);   // reused as MID bf16
    u16*   OBUF  = (u16*)  alloc((size_t)ROWS_ * D_ * 2);
    float* HEADP = (float*)alloc((size_t)51 * NSER_ * PRED_ * 4);
    u16*   MIDB  = (u16*)QKV;   // lifetime-disjoint reuse

    // weight conversion (f32 -> bf16)
    f2b_k<<<2048, 256, 0, stream>>>(w_qkv, WQKV, NL_ * D_ * 3 * D_);
    f2b_k<<<2048, 256, 0, stream>>>(w_ao,  WAO,  NL_ * D_ * D_);
    f2b_k<<<2048, 256, 0, stream>>>(w_fc,  WFC,  NL_ * D_ * 4 * D_);
    f2b_k<<<2048, 256, 0, stream>>>(w_mo,  WMO,  NL_ * 4 * D_ * D_);
    f2b_k<<<2048, 256, 0, stream>>>(w_out, WOUT, SEQ_ * D_ * PRED_);

    // decomposition + normalization
    decomp_k<<<B_, 256, 0, stream>>>(x, Z, MEANS, STDEV);
    // patch embed + prompt + wpe
    embed_k<<<dim3(NSER_, SEQ_), 256, 0, stream>>>(Z, prompt, w_in, b_in, wpe, H);

    // transformer layers
    for (int l = 0; l < NL_; ++l) {
        ln_k<<<ROWS_, 256, 0, stream>>>(H, ln1_g + l * D_, ln1_b + l * D_, ABUF);
        gemm_bf16<0, 0, 0><<<dim3(ROWS_ / 128, (3 * D_) / 128), 256, 0, stream>>>(
            ABUF, WQKV + (size_t)l * D_ * 3 * D_, b_qkv + l * 3 * D_,
            nullptr, QKV, nullptr, ROWS_, 3 * D_, D_);
        attention_k<<<dim3(NSER_, NH_), 256, 0, stream>>>(QKV, OBUF);
        gemm_bf16<1, 0, 0><<<dim3(ROWS_ / 128, D_ / 128), 256, 0, stream>>>(
            OBUF, WAO + (size_t)l * D_ * D_, b_ao + l * D_,
            H, H, nullptr, ROWS_, D_, D_);
        ln_k<<<ROWS_, 256, 0, stream>>>(H, ln2_g + l * D_, ln2_b + l * D_, ABUF);
        gemm_bf16<0, 1, 1><<<dim3(ROWS_ / 128, (4 * D_) / 128), 256, 0, stream>>>(
            ABUF, WFC + (size_t)l * D_ * 4 * D_, b_fc + l * 4 * D_,
            nullptr, nullptr, MIDB, ROWS_, 4 * D_, D_);
        gemm_bf16<1, 0, 0><<<dim3(ROWS_ / 128, D_ / 128), 256, 0, stream>>>(
            MIDB, WMO + (size_t)l * 4 * D_ * D_, b_mo + l * D_,
            H, H, nullptr, ROWS_, D_, 4 * D_);
    }

    // final LN -> bf16 (flattened = (192, 52224) head input)
    ln_k<<<ROWS_, 256, 0, stream>>>(H, lnf_g, lnf_b, ABUF);
    // head GEMM split-K
    head_gemm<<<dim3(NSER_ / 16, PRED_ / 16, 51), 32, 0, stream>>>(ABUF, WOUT, HEADP);
    // reduce + rescale + sum channels
    combine_k<<<B_, PRED_, 0, stream>>>(HEADP, b_out, MEANS, STDEV, (float*)d_out);
}
